// SlidingWindowGroupedAttention_68118181314910
// MI455X (gfx1250) — matrix operand order repaired and checked by reading, not compiled, not run on hardware
//
#include <hip/hip_runtime.h>
#include <cstdint>
#include <cstddef>

#define BB 2
#define LL 8192
#define EE 1024
#define HH 16
#define DD 64
#define WW 512
#define GG 64
#define NG (LL / GG) /* 128 */

typedef __bf16 bf16;
typedef __attribute__((ext_vector_type(8)))  __bf16 v8bf;
typedef __attribute__((ext_vector_type(16))) __bf16 v16bf;
typedef __attribute__((ext_vector_type(8)))  float  v8f;
typedef int v4i __attribute__((__vector_size__(16)));  // matches builtin param

// Optional CDNA5 async global->LDS path (guarded so compile never breaks).
#if defined(__has_builtin)
#if __has_builtin(__builtin_amdgcn_global_load_async_to_lds_b128) && \
    __has_builtin(__builtin_amdgcn_s_wait_asynccnt)
#define HAVE_ASYNC_LDS 1
#endif
#endif
#ifndef HAVE_ASYNC_LDS
#define HAVE_ASYNC_LDS 0
#endif

// ---------------------------------------------------------------------------
// WMMA helpers (CDNA5 wave32; A and B both use contiguous k = 16*(lane/16) + elem)
// ---------------------------------------------------------------------------

static __device__ inline v8f wmma_bf16(v16bf a, v16bf b, v8f c) {
  return __builtin_amdgcn_wmma_f32_16x16x32_bf16(
      /*neg_a=*/false, a, /*neg_b=*/false, b,
      /*c_mod=*/(short)0, c, /*reuse_a=*/false, /*reuse_b=*/false);
}

// A fragment: 16x32 (MxK) tile from row-major src (ld elements per row).
// Lanes 0-15: M=lane, elems 0..15 = K 0..15.
// Lanes 16-31: M=lane-16, elems 0..15 = K 16..31.
static __device__ inline v16bf frag_a(const bf16* __restrict__ src,
                                      int row0, int k0, int ld) {
  int lane = threadIdx.x & 31;
  int m  = lane & 15;
  int kg = lane >> 4;
  const bf16* p = src + (size_t)(row0 + m) * ld + k0 + kg * 16;
  v8bf lo = *(const v8bf*)(p);
  v8bf hi = *(const v8bf*)(p + 8);
  v16bf r;
#pragma unroll
  for (int e = 0; e < 8; ++e) { r[e] = lo[e]; r[e + 8] = hi[e]; }
  return r;
}

// B fragment for B[k][n] = src[(n0+n)*ld + k0+k] (W^T / K^T / V^T access).
// Lanes 0-15: N=lane, elems 0..15 = K 0..15; lanes 16-31: K 16..31.
static __device__ inline v16bf frag_b_rows(const bf16* __restrict__ src,
                                           int n0, int k0, int ld) {
  int lane = threadIdx.x & 31;
  int nn = lane & 15;
  int kh = lane >> 4;
  const bf16* p = src + (size_t)(n0 + nn) * ld + k0 + kh * 16;
  v8bf lo = *(const v8bf*)(p);
  v8bf hi = *(const v8bf*)(p + 8);
  v16bf r;
#pragma unroll
  for (int e = 0; e < 8; ++e) { r[e] = lo[e]; r[e + 8] = hi[e]; }
  return r;
}

// B fragment for keys: column n of S maps to key row min(keypos0+n, L-1).
static __device__ inline v16bf frag_b_keys(const bf16* __restrict__ src,
                                           int keypos0, int k0) {
  int lane = threadIdx.x & 31;
  int nn = lane & 15;
  int kh = lane >> 4;
  int p = keypos0 + nn;
  if (p > LL - 1) p = LL - 1;
  const bf16* q = src + (size_t)p * DD + k0 + kh * 16;
  v8bf lo = *(const v8bf*)(q);
  v8bf hi = *(const v8bf*)(q + 8);
  v16bf r;
#pragma unroll
  for (int e = 0; e < 8; ++e) { r[e] = lo[e]; r[e + 8] = hi[e]; }
  return r;
}

// ---------------------------------------------------------------------------
// fp32 -> bf16 conversion kernels
// ---------------------------------------------------------------------------

__global__ __launch_bounds__(256)
void cvt_kernel(const float* __restrict__ in, bf16* __restrict__ out, size_t n) {
  size_t i = (size_t)blockIdx.x * blockDim.x + threadIdx.x;
  size_t stride = (size_t)gridDim.x * blockDim.x;
  for (; i < n; i += stride) out[i] = (bf16)in[i];
}

// Extract value[b, 0:W, :] -> packed (B, W, E) bf16 (only first W rows used).
__global__ __launch_bounds__(256)
void cvt_v_kernel(const float* __restrict__ value, bf16* __restrict__ out) {
  size_t i = (size_t)blockIdx.x * blockDim.x + threadIdx.x;
  const size_t n = (size_t)BB * WW * EE;
  if (i >= n) return;
  size_t b = i / ((size_t)WW * EE);
  size_t rem = i % ((size_t)WW * EE);
  out[i] = (bf16)value[b * (size_t)LL * EE + rem];
}

// ---------------------------------------------------------------------------
// GEMM: C = A(MxK) * Wm^T + bias.  128x64 block tile, 8 waves, 32x32/wave.
// mode 0: out bf16 in (B, H, Lseq, D) head layout.
// mode 1: out fp32 row-major MxN.
// mode 2: out bf16 transposed per head: (B, H, D, Lseq)   [for V]
// ---------------------------------------------------------------------------

__global__ __launch_bounds__(256)
void gemm_bf16_kernel(const bf16* __restrict__ A, const bf16* __restrict__ Wm,
                      const float* __restrict__ bias, void* __restrict__ out,
                      int K, int N, int mode, int Lseq) {
  int wave = threadIdx.x >> 5;
  int row0 = blockIdx.y * 128 + (wave & 3) * 32;
  int col0 = blockIdx.x * 64 + (wave >> 2) * 32;

  v8f acc00 = {}, acc01 = {}, acc10 = {}, acc11 = {};

  // software-pipelined K loop: fetch k+32 fragments before MMA on k
  v16bf a0 = frag_a(A, row0, 0, K);
  v16bf a1 = frag_a(A, row0 + 16, 0, K);
  v16bf b0 = frag_b_rows(Wm, col0, 0, K);
  v16bf b1 = frag_b_rows(Wm, col0 + 16, 0, K);
  for (int k0 = 32; k0 < K; k0 += 32) {
    __builtin_prefetch(A + (size_t)row0 * K + k0 + 32, 0, 1);
    v16bf na0 = frag_a(A, row0, k0, K);
    v16bf na1 = frag_a(A, row0 + 16, k0, K);
    v16bf nb0 = frag_b_rows(Wm, col0, k0, K);
    v16bf nb1 = frag_b_rows(Wm, col0 + 16, k0, K);
    acc00 = wmma_bf16(a0, b0, acc00);
    acc01 = wmma_bf16(a0, b1, acc01);
    acc10 = wmma_bf16(a1, b0, acc10);
    acc11 = wmma_bf16(a1, b1, acc11);
    a0 = na0; a1 = na1; b0 = nb0; b1 = nb1;
  }
  acc00 = wmma_bf16(a0, b0, acc00);
  acc01 = wmma_bf16(a0, b1, acc01);
  acc10 = wmma_bf16(a1, b0, acc10);
  acc11 = wmma_bf16(a1, b1, acc11);

  int lane = threadIdx.x & 31;
  int mr = (lane >> 4) * 8;
  int nc = lane & 15;
#pragma unroll
  for (int mt = 0; mt < 2; ++mt) {
#pragma unroll
    for (int nt = 0; nt < 2; ++nt) {
      v8f acc = mt ? (nt ? acc11 : acc10) : (nt ? acc01 : acc00);
#pragma unroll
      for (int r = 0; r < 8; ++r) {
        int i = row0 + mt * 16 + mr + r;
        int j = col0 + nt * 16 + nc;
        float v = acc[r] + bias[j];
        if (mode == 0) {
          int b = i / Lseq, l = i % Lseq, h = j >> 6, d = j & 63;
          ((bf16*)out)[(((size_t)(b * HH + h)) * Lseq + l) * DD + d] = (bf16)v;
        } else if (mode == 1) {
          ((float*)out)[(size_t)i * N + j] = v;
        } else {
          int b = i / Lseq, l = i % Lseq, h = j >> 6, d = j & 63;
          ((bf16*)out)[(((size_t)(b * HH + h)) * DD + d) * Lseq + l] = (bf16)v;
        }
      }
    }
  }
}

// ---------------------------------------------------------------------------
// RoPE in-place on (B,H,Lseq,D) bf16; one wave32 per row, lane t = pair t.
// out[t] = xe*sin + xo*cos ; out[32+t] = -xe*cos + xo*sin
// ---------------------------------------------------------------------------

__global__ __launch_bounds__(256)
void rope_kernel(bf16* __restrict__ X, int total_rows, int Lseq) {
  int row = (int)(((size_t)blockIdx.x * blockDim.x + threadIdx.x) >> 5);
  int lane = threadIdx.x & 31;
  if (row >= total_rows) return;
  int l = row % Lseq;
  bf16* p = X + (size_t)row * DD;
  float xe = (float)p[2 * lane];
  float xo = (float)p[2 * lane + 1];
  // inv_freq = 10000^(-(2t)/64)
  float inv_freq = __expf(-((float)(2 * lane) / (float)DD) * 9.210340371976184f);
  float ang = (float)l * inv_freq;
  float s = sinf(ang), c = cosf(ang);
  bf16 o0 = (bf16)(xe * s + xo * c);
  bf16 o1 = (bf16)(-xe * c + xo * s);
  p[lane] = o0;
  p[32 + lane] = o1;
}

// ---------------------------------------------------------------------------
// Attention: one block = (b, h, group n, strip of 16 queries); 128 threads.
// S(16x512) = Qs * K^T / 8 ; masked-to-zero softmax ; ctx = P * Vt[0:512].
// Vt is per-head transposed V: (B, H, D, W).
// ---------------------------------------------------------------------------

#if HAVE_ASYNC_LDS
// Stage one 16x32-element Vt chunk (1 KB) into LDS: 2 async b128 per lane.
static __device__ inline void stage_v_chunk(const bf16* Vth, bf16 (*buf)[32],
                                            int wave, int lane, int k0) {
  const bf16* g = Vth + (size_t)(wave * 16 + (lane >> 1)) * WW
                      + k0 + (lane & 1) * 16;
  bf16* l = &buf[lane >> 1][(lane & 1) * 16];
  __builtin_amdgcn_global_load_async_to_lds_b128((v4i*)g, (v4i*)l, 0, 0);
  __builtin_amdgcn_global_load_async_to_lds_b128((v4i*)(g + 8), (v4i*)(l + 8), 0, 0);
}
#endif

__global__ __launch_bounds__(128)
void attn_kernel(const bf16* __restrict__ Q, const bf16* __restrict__ Kt,
                 const bf16* __restrict__ Vt, bf16* __restrict__ ctx) {
  __shared__ float S[16][WW];     // 32 KB
  __shared__ bf16  P[16][WW];     // 16 KB
  __shared__ float red[16][8];
#if HAVE_ASYNC_LDS
  __shared__ bf16 Vstage[2][4][16][32];  // 8 KB, double-buffered per wave
#endif

  int idx = blockIdx.x;
  int strip = idx & 3;        idx >>= 2;
  int n     = idx & (NG - 1); idx >>= 7;
  int h     = idx & (HH - 1); idx >>= 4;
  int b     = idx;

  const bf16* Qs  = Q + ((size_t)(b * HH + h) * LL + n * GG + strip * 16) * DD;
  const bf16* Kh  = Kt + (size_t)(b * HH + h) * LL * DD;
  const bf16* Vth = Vt + (size_t)(b * HH + h) * DD * WW;

  int wave = threadIdx.x >> 5;
  int lane = threadIdx.x & 31;
  int mr = (lane >> 4) * 8;
  int nc = lane & 15;

  // ---- scores: each wave owns 128 columns (8 tiles of 16x16) ----
  v16bf a0 = frag_a(Qs, 0, 0, DD);
  v16bf a1 = frag_a(Qs, 0, 32, DD);
  int kbase = n * GG;
#pragma unroll
  for (int t = 0; t < 8; ++t) {
    int c0 = wave * 128 + t * 16;
    v8f acc = {};
    acc = wmma_bf16(a0, frag_b_keys(Kh, kbase + c0, 0), acc);
    acc = wmma_bf16(a1, frag_b_keys(Kh, kbase + c0, 32), acc);
#pragma unroll
    for (int r = 0; r < 8; ++r)
      S[mr + r][c0 + nc] = acc[r] * 0.125f;   // 1/sqrt(64)
  }
  __syncthreads();

  // ---- softmax over 512 cols; invalid scores are ZERO (reference quirk) ----
  int row = threadIdx.x >> 3;
  int sub = threadIdx.x & 7;
  int nvalid = LL - kbase; if (nvalid > WW) nvalid = WW;

  float lmax = -3.0e38f;
  for (int c = sub; c < WW; c += 8) {
    float sv = (c < nvalid) ? S[row][c] : 0.0f;
    S[row][c] = sv;
    lmax = fmaxf(lmax, sv);
  }
  red[row][sub] = lmax;
  __syncthreads();
  float rmax = red[row][0];
#pragma unroll
  for (int j2 = 1; j2 < 8; ++j2) rmax = fmaxf(rmax, red[row][j2]);
  __syncthreads();

  float lsum = 0.0f;
  for (int c = sub; c < WW; c += 8) {
    float e = __expf(S[row][c] - rmax);
    S[row][c] = e;
    lsum += e;
  }
  red[row][sub] = lsum;
  __syncthreads();
  float rsum = 0.0f;
#pragma unroll
  for (int j2 = 0; j2 < 8; ++j2) rsum += red[row][j2];
  float rinv = 1.0f / rsum;
  for (int c = sub; c < WW; c += 8)
    P[row][c] = (bf16)(S[row][c] * rinv);
  __syncthreads();

  // ---- ctx strip (16x64) = P(16x512) @ V(512x64); wave owns 16 d-cols ----
  // B[k][n] = V[k][d=n] = Vt[wave*16+n][k]  (contiguous in k)
  v8f acc = {};
  const bf16* Pp = &P[0][0];
#if HAVE_ASYNC_LDS
  // Pipelined async staging: issue chunk i+1, wait for chunk i (<=2 in flight)
  stage_v_chunk(Vth, Vstage[0][wave], wave, lane, 0);
  for (int i = 0; i < 15; ++i) {
    stage_v_chunk(Vth, Vstage[(i + 1) & 1][wave], wave, lane, (i + 1) * 32);
    __builtin_amdgcn_s_wait_asynccnt(2);
    v16bf a = frag_a(Pp, 0, i * 32, WW);
    v16bf bb = frag_b_rows(&Vstage[i & 1][wave][0][0], 0, 0, 32);
    acc = wmma_bf16(a, bb, acc);
  }
  __builtin_amdgcn_s_wait_asynccnt(0);
  {
    v16bf a = frag_a(Pp, 0, 15 * 32, WW);
    v16bf bb = frag_b_rows(&Vstage[15 & 1][wave][0][0], 0, 0, 32);
    acc = wmma_bf16(a, bb, acc);
  }
#else
  for (int k0 = 0; k0 < WW; k0 += 32) {
    v16bf a = frag_a(Pp, 0, k0, WW);
    v16bf bb = frag_b_rows(Vth, wave * 16, k0, WW);
    acc = wmma_bf16(a, bb, acc);
  }
#endif
  int lrow = n * GG + strip * 16;
#pragma unroll
  for (int r = 0; r < 8; ++r) {
    int l = lrow + mr + r;
    int j = h * DD + wave * 16 + nc;
    ctx[((size_t)b * LL + l) * EE + j] = (bf16)acc[r];
  }
}

// ---------------------------------------------------------------------------
// Host launch
// ---------------------------------------------------------------------------

extern "C" void kernel_launch(void* const* d_in, const int* in_sizes, int n_in,
                              void* d_out, int out_size, void* d_ws, size_t ws_size,
                              hipStream_t stream) {
  (void)in_sizes; (void)n_in; (void)out_size; (void)ws_size;
  const float* q_in = (const float*)d_in[0];
  const float* k_in = (const float*)d_in[1];
  const float* v_in = (const float*)d_in[2];
  const float* Wq = (const float*)d_in[3];
  const float* bq = (const float*)d_in[4];
  const float* Wk = (const float*)d_in[5];
  const float* bk = (const float*)d_in[6];
  const float* Wv = (const float*)d_in[7];
  const float* bv = (const float*)d_in[8];
  const float* Wo = (const float*)d_in[9];
  const float* bo = (const float*)d_in[10];
  float* out = (float*)d_out;

  char* ws = (char*)d_ws;
  size_t off = 0;
  const size_t NQ = (size_t)BB * LL * EE;          // 16,777,216
  const size_t NW = (size_t)EE * EE;               // 1,048,576
  const size_t NV = (size_t)BB * WW * EE;          // 1,048,576
  bf16* qbf = (bf16*)(ws + off); off += NQ * 2;
  bf16* kbf = (bf16*)(ws + off); off += NQ * 2;
  bf16* vbf = (bf16*)(ws + off); off += NV * 2;
  bf16* wqb = (bf16*)(ws + off); off += NW * 2;
  bf16* wkb = (bf16*)(ws + off); off += NW * 2;
  bf16* wvb = (bf16*)(ws + off); off += NW * 2;
  bf16* wob = (bf16*)(ws + off); off += NW * 2;
  bf16* Qp  = (bf16*)(ws + off); off += NQ * 2;
  bf16* Kp  = (bf16*)(ws + off); off += NQ * 2;
  bf16* Vtp = (bf16*)(ws + off); off += (size_t)BB * HH * WW * DD * 2;
  bf16* ctx = qbf;  // qbf is dead after the Q projection; reuse for ctx

  // 1) conversions
  cvt_kernel<<<4096, 256, 0, stream>>>(q_in, qbf, NQ);
  cvt_kernel<<<4096, 256, 0, stream>>>(k_in, kbf, NQ);
  cvt_kernel<<<512, 256, 0, stream>>>(Wq, wqb, NW);
  cvt_kernel<<<512, 256, 0, stream>>>(Wk, wkb, NW);
  cvt_kernel<<<512, 256, 0, stream>>>(Wv, wvb, NW);
  cvt_kernel<<<512, 256, 0, stream>>>(Wo, wob, NW);
  cvt_v_kernel<<<(unsigned)(NV / 256), 256, 0, stream>>>(v_in, vbf);

  // 2) projections (WMMA GEMMs), 128x64 block tiles
  dim3 ggrid(EE / 64, (BB * LL) / 128);   // 16 x 128
  gemm_bf16_kernel<<<ggrid, 256, 0, stream>>>(qbf, wqb, bq, Qp, EE, EE, 0, LL);
  gemm_bf16_kernel<<<ggrid, 256, 0, stream>>>(kbf, wkb, bk, Kp, EE, EE, 0, LL);
  dim3 vgrid(EE / 64, (BB * WW) / 128);   // 16 x 8
  gemm_bf16_kernel<<<vgrid, 256, 0, stream>>>(vbf, wvb, bv, Vtp, EE, EE, 2, WW);

  // 3) RoPE on Q and K
  int rope_rows = BB * HH * LL;                       // 262144 rows
  int rope_blocks = (rope_rows * 32) / 256;           // 32768 blocks
  rope_kernel<<<rope_blocks, 256, 0, stream>>>(Qp, rope_rows, LL);
  rope_kernel<<<rope_blocks, 256, 0, stream>>>(Kp, rope_rows, LL);

  // 4) attention
  attn_kernel<<<BB * HH * NG * 4, 128, 0, stream>>>(Qp, Kp, Vtp, ctx);

  // 5) output projection -> fp32
  gemm_bf16_kernel<<<ggrid, 256, 0, stream>>>(ctx, wob, bo, out, EE, EE, 1, LL);
}
